// GPTAttention_50594714747473
// MI455X (gfx1250) — compile-verified, matrix operand order checked by reading, not run on hardware
//
#include <hip/hip_runtime.h>
#include <hip/hip_bf16.h>

#define BATCH 4
#define SEQ   2048
#define HID   1024
#define NHEAD 16
#define HDIM  64

typedef __bf16 bf16_t;
typedef __attribute__((ext_vector_type(16))) __bf16 v16bf;
typedef __attribute__((ext_vector_type(8)))  __bf16 v8bf;
typedef __attribute__((ext_vector_type(4)))  __bf16 v4bf;
typedef __attribute__((ext_vector_type(8)))  float  v8f;
typedef __attribute__((ext_vector_type(4)))  float  f32x4;
typedef int v4i32 __attribute__((vector_size(16)));

// LDS row strides (elements). Row pitch multiple of 16 B (ds_load_b128)
// AND conflict-free across the 16 lanes of a half-wave:
//   56 elts = 112 B : m*28 mod 64 -> 16 distinct banks
//   72 elts = 144 B : m*36 mod 64 -> 16 distinct banks
#define LD32 56
#define LD64 72

#define HAS_ASYNC_LDS __has_builtin(__builtin_amdgcn_global_load_async_to_lds_b128)

// 16B global->LDS copy: async (ASYNCcnt, no VGPR round trip) on gfx1250
// toolchains that expose the builtin; synchronous vector copy otherwise.
// Builtin signature (from compiler diagnostic): (v4i32*, v4i32*, int, int).
__device__ __forceinline__ void copy16_g2l(bf16_t* l, const uint4* g) {
#if HAS_ASYNC_LDS
  __builtin_amdgcn_global_load_async_to_lds_b128(
      (v4i32*)const_cast<uint4*>(g), (v4i32*)l, 0, 0);
#else
  *(uint4*)l = *g;
#endif
}

// Wait for outstanding async tile loads, then workgroup barrier.
__device__ __forceinline__ void staging_fence() {
#if HAS_ASYNC_LDS
#if __has_builtin(__builtin_amdgcn_s_wait_asynccnt)
  __builtin_amdgcn_s_wait_asynccnt(0);
#else
  asm volatile("s_wait_asynccnt 0x0" ::: "memory");
#endif
#endif
  __syncthreads();
}

__device__ __forceinline__ v8f zero8() {
  v8f z;
#pragma unroll
  for (int i = 0; i < 8; ++i) z[i] = 0.0f;
  return z;
}

__device__ __forceinline__ v8f wmma_bf16(v16bf a, v16bf b, v8f c) {
  // D = A(16x32) * B(32x16) + C, f32 accumulate
  return __builtin_amdgcn_wmma_f32_16x16x32_bf16(false, a, false, b, (short)0, c,
                                                 false, false);
}

// 16-bit operand fragment (16x32 A-layout, ISA 7.12.2) from a row-major
// source with 16B-aligned rows: two contiguous ds_load_b128 per fragment.
__device__ __forceinline__ v16bf load_frag(const bf16_t* src, int ld) {
  const int lane = threadIdx.x & 31;
  const int m    = lane & 15;
  const int kb   = (lane >> 4) << 3;
  const bf16_t* r = src + (size_t)m * ld + kb;
  v8bf lo = *(const v8bf*)(r);        // K = kb .. kb+7
  v8bf hi = *(const v8bf*)(r + 16);   // K = kb+16 .. kb+23
  return __builtin_shufflevector(lo, hi, 0, 1, 2, 3, 4, 5, 6, 7,
                                 8, 9, 10, 11, 12, 13, 14, 15);
}

// ---------------------------------------------------------------------------
// Pre-pass A: streaming f32 -> bf16 conversion (layout preserved).
// ---------------------------------------------------------------------------
__global__ __launch_bounds__(256)
void cvt_bf16_kernel(const float* __restrict__ in, bf16_t* __restrict__ out) {
  size_t i = (size_t)blockIdx.x * 256 + threadIdx.x;
  f32x4 x = *(const f32x4*)&in[i * 4];
  v4bf y;
#pragma unroll
  for (int j = 0; j < 4; ++j) y[j] = (bf16_t)x[j];
  *(v4bf*)&out[i * 4] = y;
}

// ---------------------------------------------------------------------------
// Pre-pass B: f32 [R][C] -> bf16 [C][R] transpose+convert, done ONCE so the
// GEMM k-loops never transpose. 64x64 tiles via LDS; both sides coalesced.
// ---------------------------------------------------------------------------
__global__ __launch_bounds__(128)
void transpose_cvt_kernel(const float* __restrict__ in, bf16_t* __restrict__ out,
                          int R, int C) {
  __shared__ __align__(16) bf16_t T[64][LD64];  // [c][r]
  const int tid = threadIdx.x;
  const int r0  = blockIdx.x * 64;
  const int c0  = blockIdx.y * 64;
#pragma unroll
  for (int it = 0; it < 8; ++it) {
    int i = tid + it * 128;  // 1024 float4 = 64 rows x 16 float4
    int r = i >> 4, c4 = i & 15;
    f32x4 x = *(const f32x4*)&in[(size_t)(r0 + r) * C + c0 + 4 * c4];
#pragma unroll
    for (int j = 0; j < 4; ++j) T[4 * c4 + j][r] = (bf16_t)x[j];
  }
  __syncthreads();
#pragma unroll
  for (int it = 0; it < 4; ++it) {
    int i = tid + it * 128;  // 512 uint4 = 64 out-rows x 8 uint4
    int c = i >> 3, r8 = i & 7;
    *(uint4*)&out[(size_t)(c0 + c) * R + r0 + 8 * r8] = *(const uint4*)&T[c][8 * r8];
  }
}

// ---------------------------------------------------------------------------
// Kernel 1: fused QKV projection, all-bf16 operands, DOUBLE-BUFFERED LDS:
// tile k+1 streams in (async-to-LDS when available) while tile k is consumed
// by WMMAs; one asynccnt-wait + one barrier per k-step.
// Xb[8192,1024] bf16 x Wt[3072,1024] bf16 ([n][k], pre-transposed).
// q,k -> token-major [B, NH, S, HD] (q pre-scaled by 1/8).
// v   -> d-major    [B, NH, HD, S] (packed 16B epilogue stores).
// WG 128 thr (4 waves); WG tile 64x128; wave tile 32x64 (2x4 WMMA tiles).
// ---------------------------------------------------------------------------
__global__ __launch_bounds__(128)
void qkv_gemm_kernel(const bf16_t* __restrict__ Xb, const bf16_t* __restrict__ Wt,
                     bf16_t* __restrict__ qws, bf16_t* __restrict__ kws,
                     bf16_t* __restrict__ vws) {
  __shared__ __align__(16) bf16_t As[2][64][LD32];    // [buf][row][K]
  __shared__ __align__(16) bf16_t Bst[2][128][LD32];  // [buf][col][K]

  const int tid  = threadIdx.x;
  const int wave = tid >> 5;
  const int row0 = blockIdx.x * 64;
  const int col0 = blockIdx.y * 128;
  const int wr   = (wave >> 1) * 32;  // 0 or 32
  const int wc   = (wave & 1) * 64;   // 0 or 64

  const uint4* ag = (const uint4*)Xb;  // 8 bf16 per uint4
  const uint4* wg = (const uint4*)Wt;

  auto stage = [&](int buf, int k0) {
#pragma unroll
    for (int it = 0; it < 2; ++it) {   // A tile 64x32: 256 x 16B
      int i = tid + it * 128;
      int r = i >> 2, c = i & 3;
      copy16_g2l(&As[buf][r][8 * c],
                 &ag[(size_t)(row0 + r) * (HID / 8) + (k0 >> 3) + c]);
    }
#pragma unroll
    for (int it = 0; it < 4; ++it) {   // B tile 128x32: 512 x 16B
      int i = tid + it * 128;
      int r = i >> 2, c = i & 3;
      copy16_g2l(&Bst[buf][r][8 * c],
                 &wg[(size_t)(col0 + r) * (HID / 8) + (k0 >> 3) + c]);
    }
  };

  v8f acc[2][4];
#pragma unroll
  for (int i = 0; i < 2; ++i)
#pragma unroll
    for (int j = 0; j < 4; ++j) acc[i][j] = zero8();

  stage(0, 0);
  staging_fence();
  int buf = 0;
  for (int k0 = 0; k0 < HID; k0 += 32) {
    if (k0 + 32 < HID) stage(buf ^ 1, k0 + 32);  // overlap with compute

    v16bf a0 = load_frag(&As[buf][wr][0], LD32);
    v16bf a1 = load_frag(&As[buf][wr + 16][0], LD32);
#pragma unroll
    for (int tj = 0; tj < 4; ++tj) {
      v16bf bfr = load_frag(&Bst[buf][wc + 16 * tj][0], LD32);
      acc[0][tj] = wmma_bf16(a0, bfr, acc[0][tj]);
      acc[1][tj] = wmma_bf16(a1, bfr, acc[1][tj]);
    }
    staging_fence();
    buf ^= 1;
  }

  // Epilogue: tile-uniform head/batch decode hoisted out of the lane loop.
  const int lane = tid & 31;
  const int n    = lane & 15;
  const int hi   = lane >> 4;
#pragma unroll
  for (int tj = 0; tj < 4; ++tj) {
    int cbase = col0 + wc + tj * 16;      // 16-wide span, never crosses head
    int which = cbase >> 10;              // 0=q 1=k 2=v (uniform per tile)
    int hc    = cbase & 1023;
    int head  = hc >> 6;
    int dcol  = (hc & 63) + n;
    if (which == 2) {
      // V d-major: lane's 8 rows = 8 consecutive tokens -> one 16B store.
#pragma unroll
      for (int ti = 0; ti < 2; ++ti) {
        int rbase = row0 + wr + ti * 16 + hi * 8;
        int bb = rbase >> 11;
        int ss = rbase & 2047;            // multiple of 8 -> 16B aligned
        size_t base = (((size_t)bb * NHEAD + head) * HDIM + dcol) * SEQ + ss;
        v8bf y;
#pragma unroll
        for (int v = 0; v < 8; ++v) y[v] = (bf16_t)acc[ti][tj][v];
        *(v8bf*)&vws[base] = y;
      }
    } else {
      bf16_t* dst  = (which == 0) ? qws : kws;
      float   scal = (which == 0) ? 0.125f : 1.0f;  // 1/sqrt(64) folded into q
#pragma unroll
      for (int ti = 0; ti < 2; ++ti) {
        int rbase = row0 + wr + ti * 16 + hi * 8;   // +v; never crosses batch
        int bb = rbase >> 11;
        int ss = rbase & 2047;
        size_t base = ((((size_t)bb * NHEAD + head) * SEQ) + ss) * HDIM + dcol;
#pragma unroll
        for (int v = 0; v < 8; ++v)
          dst[base + (size_t)v * HDIM] = (bf16_t)(acc[ti][tj][v] * scal);
      }
    }
  }
}

// ---------------------------------------------------------------------------
// Kernel 2: causal flash attention, double-buffered K/V staging.
// One WG per (b, h, 64-query tile); 4 waves x 16 query rows. K staged
// row-major (it IS the B operand for Q.K^T); V arrives d-major from
// kernel 1 -> both tiles stage as pure 16B copies (async when available).
// Online softmax in registers; P relaid via wave-private LDS.
// Output bf16 to [B, S, NH, HD].
// ---------------------------------------------------------------------------
__global__ __launch_bounds__(128)
void flash_attn_kernel(const bf16_t* __restrict__ qws, const bf16_t* __restrict__ kws,
                       const bf16_t* __restrict__ vws, bf16_t* __restrict__ aout) {
  __shared__ __align__(16) bf16_t Ks[2][64][LD64];  // [buf][key][d]
  __shared__ __align__(16) bf16_t Vt[2][64][LD64];  // [buf][d][key]
  __shared__ __align__(16) bf16_t Ps[4][16][LD64];  // wave-private P tiles

  const int tid  = threadIdx.x;
  const int wave = tid >> 5;
  const int lane = tid & 31;
  const int n    = lane & 15;
  const int hi   = lane >> 4;

  const int qt = blockIdx.x & 31;  // 32 query tiles of 64 rows
  const int bh = blockIdx.x >> 5;  // b*16 + h
  const int b  = bh >> 4;
  const int h  = bh & 15;

  const bf16_t* qp = qws + (size_t)bh * SEQ * HDIM;
  const bf16_t* kp = kws + (size_t)bh * SEQ * HDIM;
  const bf16_t* vp = vws + (size_t)bh * HDIM * SEQ;  // d-major!

  const int qr0 = qt * 64 + wave * 16;

  // Q fragments straight from global (rows contiguous, 64B-aligned).
  v16bf qa0 = load_frag(qp + (size_t)qr0 * HDIM, HDIM);
  v16bf qa1 = load_frag(qp + (size_t)qr0 * HDIM + 32, HDIM);

  auto stageKV = [&](int buf, int kb) {
    const uint4* kg = (const uint4*)(kp + (size_t)kb * 64 * HDIM);
    const uint4* vg = (const uint4*)vp;
#pragma unroll
    for (int it = 0; it < 4; ++it) {
      int i = tid + it * 128;
      int r = i >> 3, c = i & 7;
      copy16_g2l(&Ks[buf][r][8 * c], &kg[r * 8 + c]);
      copy16_g2l(&Vt[buf][r][8 * c],
                 &vg[(size_t)r * (SEQ / 8) + (size_t)kb * 8 + c]);
    }
  };

  v8f   O[4];
  float m_run[8], l_run[8];
#pragma unroll
  for (int t = 0; t < 4; ++t) O[t] = zero8();
#pragma unroll
  for (int v = 0; v < 8; ++v) { m_run[v] = -3.0e38f; l_run[v] = 0.0f; }

  stageKV(0, 0);
  staging_fence();
  int buf = 0;
  for (int kb = 0; kb <= qt; ++kb) {
    if (kb < qt) stageKV(buf ^ 1, kb + 1);  // overlap with compute

    // Scores S = Q x K^T : row-major K tile IS the B operand (lane = key col).
    v8f sc[4];
#pragma unroll
    for (int t = 0; t < 4; ++t) {
      v16bf kf0 = load_frag(&Ks[buf][t * 16][0], LD64);   // d = 0..31
      v16bf kf1 = load_frag(&Ks[buf][t * 16][32], LD64);  // d = 32..63
      v8f s = zero8();
      s = wmma_bf16(qa0, kf0, s);
      s = wmma_bf16(qa1, kf1, s);
      sc[t] = s;
    }

    // Causal mask on the diagonal block only.
    if (kb == qt) {
#pragma unroll
      for (int t = 0; t < 4; ++t)
#pragma unroll
        for (int v = 0; v < 8; ++v) {
          int qg   = qr0 + v + hi * 8;
          int kcol = kb * 64 + t * 16 + n;
          if (kcol > qg) sc[t][v] = -3.0e38f;
        }
    }

    // Online softmax. Row r = v + 8*hi lives in 16 lanes of one half-wave;
    // xor masks 1/2/4/8 reduce within the half (bit4 untouched, wave32-safe).
#pragma unroll
    for (int v = 0; v < 8; ++v) {
      float rmax = fmaxf(fmaxf(sc[0][v], sc[1][v]), fmaxf(sc[2][v], sc[3][v]));
#pragma unroll
      for (int off = 8; off >= 1; off >>= 1)
        rmax = fmaxf(rmax, __shfl_xor(rmax, off, 32));
      float mnew  = fmaxf(m_run[v], rmax);
      float alpha = __expf(m_run[v] - mnew);
      m_run[v] = mnew;
      float rsum = 0.0f;
#pragma unroll
      for (int t = 0; t < 4; ++t) {
        float p = __expf(sc[t][v] - mnew);
        sc[t][v] = p;
        rsum += p;
      }
#pragma unroll
      for (int off = 8; off >= 1; off >>= 1)
        rsum += __shfl_xor(rsum, off, 32);
      l_run[v] = l_run[v] * alpha + rsum;
#pragma unroll
      for (int t = 0; t < 4; ++t) O[t][v] *= alpha;
    }

    // Relay P (C/D layout) -> A-fragment layout via wave-private LDS.
    // DS ops are in-order within a wave; no workgroup barrier needed here.
#pragma unroll
    for (int t = 0; t < 4; ++t)
#pragma unroll
      for (int v = 0; v < 8; ++v)
        Ps[wave][v + hi * 8][t * 16 + n] = (bf16_t)sc[t][v];

    v16bf pa0 = load_frag(&Ps[wave][0][0], LD64);   // keys 0..31
    v16bf pa1 = load_frag(&Ps[wave][0][32], LD64);  // keys 32..63

#pragma unroll
    for (int t = 0; t < 4; ++t) {
      v16bf vb0 = load_frag(&Vt[buf][t * 16][0], LD64);   // keys 0..31
      v16bf vb1 = load_frag(&Vt[buf][t * 16][32], LD64);  // keys 32..63
      O[t] = wmma_bf16(pa0, vb0, O[t]);
      O[t] = wmma_bf16(pa1, vb1, O[t]);
    }
    staging_fence();  // async tiles for kb+1 landed; cur buf free to reuse
    buf ^= 1;
  }

  // Normalize (8 reciprocals, not 32 divides) and store bf16 to [B,S,NH,HD].
  float linv[8];
#pragma unroll
  for (int v = 0; v < 8; ++v) linv[v] = 1.0f / l_run[v];
#pragma unroll
  for (int t = 0; t < 4; ++t) {
    size_t base =
        (((size_t)b * SEQ + qr0 + hi * 8) * NHEAD + h) * HDIM + t * 16 + n;
#pragma unroll
    for (int v = 0; v < 8; ++v)
      aout[base + (size_t)v * (NHEAD * HDIM)] = (bf16_t)(O[t][v] * linv[v]);
  }
}

// ---------------------------------------------------------------------------
// Kernel 3: output projection, all-bf16 operands, double-buffered LDS.
// A = attn_out bf16 [8192,1024] row-major, Wt = w_out bf16 [N][K] (pre-
// transposed) -> out f32. Same 64x128 WG / 32x64 wave tiling.
// ---------------------------------------------------------------------------
__global__ __launch_bounds__(128)
void out_proj_kernel(const bf16_t* __restrict__ A, const bf16_t* __restrict__ Wt,
                     float* __restrict__ out) {
  __shared__ __align__(16) bf16_t As[2][64][LD32];
  __shared__ __align__(16) bf16_t Bst[2][128][LD32];

  const int tid  = threadIdx.x;
  const int wave = tid >> 5;
  const int row0 = blockIdx.x * 64;
  const int col0 = blockIdx.y * 128;
  const int wr   = (wave >> 1) * 32;
  const int wc   = (wave & 1) * 64;

  const uint4* ag = (const uint4*)A;
  const uint4* wg = (const uint4*)Wt;

  auto stage = [&](int bufi, int k0) {
#pragma unroll
    for (int it = 0; it < 2; ++it) {
      int i = tid + it * 128;
      int r = i >> 2, c = i & 3;
      copy16_g2l(&As[bufi][r][8 * c],
                 &ag[(size_t)(row0 + r) * (HID / 8) + (k0 >> 3) + c]);
    }
#pragma unroll
    for (int it = 0; it < 4; ++it) {
      int i = tid + it * 128;
      int r = i >> 2, c = i & 3;
      copy16_g2l(&Bst[bufi][r][8 * c],
                 &wg[(size_t)(col0 + r) * (HID / 8) + (k0 >> 3) + c]);
    }
  };

  v8f acc[2][4];
#pragma unroll
  for (int i = 0; i < 2; ++i)
#pragma unroll
    for (int j = 0; j < 4; ++j) acc[i][j] = zero8();

  stage(0, 0);
  staging_fence();
  int buf = 0;
  for (int k0 = 0; k0 < HID; k0 += 32) {
    if (k0 + 32 < HID) stage(buf ^ 1, k0 + 32);

    v16bf a0 = load_frag(&As[buf][wr][0], LD32);
    v16bf a1 = load_frag(&As[buf][wr + 16][0], LD32);
#pragma unroll
    for (int tj = 0; tj < 4; ++tj) {
      v16bf bfr = load_frag(&Bst[buf][wc + 16 * tj][0], LD32);
      acc[0][tj] = wmma_bf16(a0, bfr, acc[0][tj]);
      acc[1][tj] = wmma_bf16(a1, bfr, acc[1][tj]);
    }
    staging_fence();
    buf ^= 1;
  }

  const int lane = tid & 31;
  const int n    = lane & 15;
  const int hi   = lane >> 4;
#pragma unroll
  for (int tj = 0; tj < 4; ++tj) {
    int gcol = col0 + wc + tj * 16 + n;
#pragma unroll
    for (int ti = 0; ti < 2; ++ti) {
      int rbase = row0 + wr + ti * 16 + hi * 8;
#pragma unroll
      for (int v = 0; v < 8; ++v)
        out[(size_t)(rbase + v) * HID + gcol] = acc[ti][tj][v];
    }
  }
}

// ---------------------------------------------------------------------------
extern "C" void kernel_launch(void* const* d_in, const int* in_sizes, int n_in,
                              void* d_out, int out_size, void* d_ws, size_t ws_size,
                              hipStream_t stream) {
  const float* X    = (const float*)d_in[0];  // hidden_states [4,2048,1024]
  const float* Win  = (const float*)d_in[1];  // w_in  [1024,3072]
  const float* Wout = (const float*)d_in[2];  // w_out [1024,1024]
  float*       out  = (float*)d_out;          // [4,2048,1024] f32

  const size_t per = (size_t)BATCH * NHEAD * SEQ * HDIM;  // 8,388,608 elems
  bf16_t* qws   = (bf16_t*)d_ws;
  bf16_t* kws   = qws + per;
  bf16_t* vws   = kws + per;          // d-major [B,NH,HD,S]
  bf16_t* aout  = vws + per;
  bf16_t* Xb    = aout + per;         // X as bf16, row-major [8192,1024]
  bf16_t* WtIn  = Xb + per;           // w_in^T  bf16 [3072,1024]
  bf16_t* WtOut = WtIn + (size_t)3 * HID * HID;  // w_out^T bf16 [1024,1024]
  // total ~88 MB bf16 workspace

  // One-time precision/layout pre-passes (stream-ordered).
  cvt_bf16_kernel<<<dim3((unsigned)(per / 4 / 256)), 256, 0, stream>>>(X, Xb);
  transpose_cvt_kernel<<<dim3(HID / 64, 3 * HID / 64), 128, 0, stream>>>(
      Win, WtIn, HID, 3 * HID);
  transpose_cvt_kernel<<<dim3(HID / 64, HID / 64), 128, 0, stream>>>(
      Wout, WtOut, HID, HID);

  dim3 g1(BATCH * SEQ / 64, 3 * HID / 128);  // 128 x 24
  qkv_gemm_kernel<<<g1, 128, 0, stream>>>(Xb, WtIn, qws, kws, vws);

  dim3 g2(BATCH * NHEAD * (SEQ / 64));       // 2048
  flash_attn_kernel<<<g2, 128, 0, stream>>>(qws, kws, vws, aout);

  dim3 g3(BATCH * SEQ / 64, HID / 128);      // 128 x 8
  out_proj_kernel<<<g3, 128, 0, stream>>>(aout, WtOut, out);
}
